// S6Layer_91122026152035
// MI455X (gfx1250) — matrix operand order repaired and checked by reading, not compiled, not run on hardware
//
#include <hip/hip_runtime.h>
#include <hip/hip_fp16.h>

typedef __attribute__((ext_vector_type(16))) _Float16 v16h;
typedef __attribute__((ext_vector_type(8)))  _Float16 v8h;
typedef __attribute__((ext_vector_type(8)))  float    v8f;

// native int4 vector type matching the async-LDS builtin's pointee type
typedef int v4i_native __attribute__((vector_size(16)));
typedef __attribute__((address_space(1))) v4i_native* as1_v4i_ptr;
typedef __attribute__((address_space(3))) v4i_native* as3_v4i_ptr;

#define B_    4
#define S_    4096
#define DM    1024
#define DS    16
#define M_TOT (B_*S_)          // 16384
#define CHUNK 128
#define NCHUNK (S_/CHUNK)      // 32
#define NCHAIN (B_*DS)         // 64

#ifndef __has_builtin
#define __has_builtin(x) 0
#endif

#if __has_builtin(__builtin_amdgcn_global_load_async_to_lds_b128)
#define HAVE_ASYNC_LDS 1
#else
#define HAVE_ASYNC_LDS 0
#endif

// stage 16 bytes global -> LDS (async if the toolchain exposes it)
__device__ __forceinline__ void stage_b128(const _Float16* g, _Float16* l) {
#if HAVE_ASYNC_LDS
  __builtin_amdgcn_global_load_async_to_lds_b128(
      (as1_v4i_ptr)g, (as3_v4i_ptr)l, 0, 0);
#else
  *(v8h*)l = *(const v8h*)g;
#endif
}

__device__ __forceinline__ void stage_wait() {
#if HAVE_ASYNC_LDS
#if __has_builtin(__builtin_amdgcn_s_wait_asynccnt)
  __builtin_amdgcn_s_wait_asynccnt(0);
#else
  asm volatile("s_wait_asynccnt 0x0" ::: "memory");
#endif
#endif
}

// ---------------------------------------------------------------- converts
__global__ __launch_bounds__(256) void cvt_f32_f16_kernel(
    const float* __restrict__ in, _Float16* __restrict__ out, int n) {
  int i = blockIdx.x * 256 + threadIdx.x;
  if (i < n) out[i] = (_Float16)in[i];
}

// per-row convert x -> f16 and compute row mean
__global__ __launch_bounds__(256) void cvt_x_kernel(
    const float* __restrict__ x, _Float16* __restrict__ xh,
    float* __restrict__ xmean) {
  __shared__ float red[256];
  int m = blockIdx.x;
  const float* row = x + (size_t)m * DM;
  _Float16* orow = xh + (size_t)m * DM;
  float s = 0.f;
  for (int c = threadIdx.x; c < DM; c += 256) {
    float v = row[c];
    orow[c] = (_Float16)v;
    s += v;
  }
  red[threadIdx.x] = s;
  __syncthreads();
  for (int off = 128; off > 0; off >>= 1) {
    if (threadIdx.x < off) red[threadIdx.x] += red[threadIdx.x + off];
    __syncthreads();
  }
  if (threadIdx.x == 0) xmean[m] = red[0] * (1.0f / DM);
}

// ---------------------------------------------------------------- WMMA frags
// A-matrix 16x32 f16, contiguous K order: lanes 0-15 -> K {0..15},
// lanes 16-31 -> K {16..31}; row M = lane & 15.
__device__ __forceinline__ v16h load_a_frag(const _Float16* p, int half) {
  const _Float16* q = p + half * 16;
  v8h c0 = *(const v8h*)(q);
  v8h c1 = *(const v8h*)(q + 8);
  return __builtin_shufflevector(c0, c1, 0,1,2,3,4,5,6,7,8,9,10,11,12,13,14,15);
}

__device__ __forceinline__ float sigmoidf_fast(float y) {
  return 1.0f / (1.0f + __expf(-y));
}

// ---------------------------------------------------------------- delta GEMM
// Y = Xh (16384x1024) * W_delta^T (1024x1024), fused sigmoid row-sum.
// Block = 8 waves stacked in M (tile 128M x 64N). Grid (16 Nblk, 128 Mblk).
// B tile (64 cols x 32 K, f16 = 4 KB) is staged to LDS with async
// global->LDS copies, double-buffered across the K loop.
__global__ __launch_bounds__(256) void delta_gemm_kernel(
    const _Float16* __restrict__ Xh, const _Float16* __restrict__ Wdh,
    const float* __restrict__ b_delta, float* __restrict__ delta_acc) {
  __shared__ alignas(32) _Float16 bbuf[2][64 * 32];

  const int t    = threadIdx.x;
  const int lane = t & 31;
  const int wave = t >> 5;
  const int r    = lane & 15;
  const int half = lane >> 4;
  const int nbase = blockIdx.x * 64;
  const int mrow  = blockIdx.y * 128 + wave * 16;

  const _Float16* arow = Xh + (size_t)(mrow + r) * DM;

  // producer mapping: 256 threads x 16B = one 4 KB tile per step
  const int nload = t >> 2;          // 0..63 : column of W tile
  const int kq    = (t & 3) * 8;     // 0,8,16,24 : K offset in halves
  const _Float16* wsrc = Wdh + (size_t)(nbase + nload) * DM + kq;

  // prologue: stage K-chunk 0 into buffer 0
  stage_b128(wsrc, &bbuf[0][nload * 32 + kq]);

  v8f c0 = {}, c1 = {}, c2 = {}, c3 = {};
  int cur = 0;
  for (int k = 0; k < DM; k += 32) {
    stage_wait();        // my async writes to bbuf[cur] have landed
    __syncthreads();     // all writes landed; all reads of bbuf[cur^1] done
    if (k + 32 < DM)     // uniform branch: EXEC stays all-ones
      stage_b128(wsrc + k + 32, &bbuf[cur ^ 1][nload * 32 + kq]);

    __builtin_prefetch(arow + k + 128, 0, 0);
    v16h a  = load_a_frag(arow + k, half);
    v16h b0 = *(const v16h*)&bbuf[cur][(0 * 16 + r) * 32 + half * 16];
    v16h b1 = *(const v16h*)&bbuf[cur][(1 * 16 + r) * 32 + half * 16];
    v16h b2 = *(const v16h*)&bbuf[cur][(2 * 16 + r) * 32 + half * 16];
    v16h b3 = *(const v16h*)&bbuf[cur][(3 * 16 + r) * 32 + half * 16];
    c0 = __builtin_amdgcn_wmma_f32_16x16x32_f16(false, a, false, b0, (short)0, c0, false, false);
    c1 = __builtin_amdgcn_wmma_f32_16x16x32_f16(false, a, false, b1, (short)0, c1, false, false);
    c2 = __builtin_amdgcn_wmma_f32_16x16x32_f16(false, a, false, b2, (short)0, c2, false, false);
    c3 = __builtin_amdgcn_wmma_f32_16x16x32_f16(false, a, false, b3, (short)0, c3, false, false);
    cur ^= 1;
  }

  // epilogue: + b_delta, sigmoid, sum across the 64 N columns of this block
  union { v8f v; float f[8]; } u0, u1, u2, u3;
  u0.v = c0; u1.v = c1; u2.v = c2; u3.v = c3;
  const float bd0 = b_delta[nbase + 0  + r];
  const float bd1 = b_delta[nbase + 16 + r];
  const float bd2 = b_delta[nbase + 32 + r];
  const float bd3 = b_delta[nbase + 48 + r];
  float sums[8];
  #pragma unroll
  for (int i = 0; i < 8; ++i) {
    float s = sigmoidf_fast(u0.f[i] + bd0);
    s += sigmoidf_fast(u1.f[i] + bd1);
    s += sigmoidf_fast(u2.f[i] + bd2);
    s += sigmoidf_fast(u3.f[i] + bd3);
    sums[i] = s;
  }
  // reduce across the 16 lanes of each half (columns); rows differ per half
  #pragma unroll
  for (int i = 0; i < 8; ++i) {
    #pragma unroll
    for (int m = 8; m >= 1; m >>= 1)
      sums[i] += __shfl_xor(sums[i], m, 16);
  }
  if (r == 0) {
    #pragma unroll
    for (int i = 0; i < 8; ++i)
      atomicAdd(&delta_acc[mrow + half * 8 + i], sums[i]);
  }
}

// ---------------------------------------------------------------- B GEMM
// Bm = Xh (16384x1024) * W_B^T (1024x16) + b_B. One wave per 16-row M tile.
// W_B is 32 KB -> lives in L2; direct global B-fragment loads are fine.
__global__ __launch_bounds__(256) void bm_gemm_kernel(
    const _Float16* __restrict__ Xh, const _Float16* __restrict__ Wbh,
    const float* __restrict__ b_B, float* __restrict__ Bm) {
  const int lane = threadIdx.x & 31;
  const int wave = threadIdx.x >> 5;
  const int r    = lane & 15;
  const int half = lane >> 4;
  const int mrow = blockIdx.x * 128 + wave * 16;

  const _Float16* arow = Xh  + (size_t)(mrow + r) * DM;
  const _Float16* brow = Wbh + (size_t)r * DM + half * 16;

  v8f c = {};
  for (int k = 0; k < DM; k += 32) {
    v16h a = load_a_frag(arow + k, half);
    v16h b = *(const v16h*)(brow + k);
    c = __builtin_amdgcn_wmma_f32_16x16x32_f16(false, a, false, b, (short)0, c, false, false);
  }
  union { v8f v; float f[8]; } u; u.v = c;
  const float bb = b_B[r];
  #pragma unroll
  for (int i = 0; i < 8; ++i) {
    int row = mrow + half * 8 + i;
    Bm[(size_t)row * DS + r] = u.f[i] + bb;
  }
}

// ---------------------------------------------------------------- pointwise
__global__ __launch_bounds__(256) void abar_bu_kernel(
    const float* __restrict__ delta_acc, const float* __restrict__ Bm,
    const float* __restrict__ xmean, const float* __restrict__ A_log,
    float* __restrict__ A_bar, float* __restrict__ Bu) {
  int idx = blockIdx.x * 256 + threadIdx.x;   // over M_TOT*DS
  int m = idx >> 4;
  int n = idx & 15;
  float delta = 0.001f + 0.099f * (delta_acc[m] * (1.0f / DM));
  float A = -__expf(A_log[n]);
  A_bar[idx] = __expf(delta * A);
  Bu[idx]    = delta * Bm[idx] * xmean[m];
}

// ---------------------------------------------------------------- scan (3 phases)
__global__ __launch_bounds__(256) void scan_phase1_kernel(
    const float* __restrict__ A_bar, const float* __restrict__ Bu,
    float* __restrict__ P, float* __restrict__ Q) {
  int t = blockIdx.x * 256 + threadIdx.x;     // 0..NCHAIN*NCHUNK-1
  int chain = t / NCHUNK, chunk = t % NCHUNK;
  int b = chain >> 4, n = chain & 15;
  size_t base = ((size_t)b * S_ + (size_t)chunk * CHUNK) * DS + n;
  float p = 1.f, q = 0.f;
  for (int i = 0; i < CHUNK; ++i) {
    float a  = A_bar[base + (size_t)i * DS];
    float bu = Bu   [base + (size_t)i * DS];
    p *= a;
    q = a * q + bu;
  }
  P[t] = p; Q[t] = q;
}

__global__ void scan_phase2_kernel(const float* __restrict__ P,
                                   const float* __restrict__ Q,
                                   float* __restrict__ Init) {
  int chain = threadIdx.x;                    // blockDim = NCHAIN
  float s = 0.f;
  for (int c = 0; c < NCHUNK; ++c) {
    int t = chain * NCHUNK + c;
    Init[t] = s;
    s = P[t] * s + Q[t];
  }
}

__global__ __launch_bounds__(256) void scan_phase3_kernel(
    const float* __restrict__ A_bar, const float* __restrict__ Bu,
    const float* __restrict__ Init, float* __restrict__ states) {
  int t = blockIdx.x * 256 + threadIdx.x;
  int chain = t / NCHUNK, chunk = t % NCHUNK;
  int b = chain >> 4, n = chain & 15;
  size_t base = ((size_t)b * S_ + (size_t)chunk * CHUNK) * DS + n;
  float s = Init[t];
  for (int i = 0; i < CHUNK; ++i) {
    size_t o = base + (size_t)i * DS;
    s = A_bar[o] * s + Bu[o];
    states[o] = s;
  }
}

// ---------------------------------------------------------------- output
// out[m][d] = sum_n states[m][n] * W_out[d][n] + b_out[d] + D_param[d]*x[m][d]
__global__ __launch_bounds__(256) void out_gemm_kernel(
    const float* __restrict__ states, const float* __restrict__ W_out,
    const float* __restrict__ b_out, const float* __restrict__ D_param,
    const float* __restrict__ x, float* __restrict__ out) {
  __shared__ float st[DS];
  int m = blockIdx.x >> 2;
  int d = ((blockIdx.x & 3) << 8) + threadIdx.x;
  if (threadIdx.x < DS) st[threadIdx.x] = states[(size_t)m * DS + threadIdx.x];
  __syncthreads();
  float acc = b_out[d] + D_param[d] * x[(size_t)m * DM + d];
  const float* w = W_out + (size_t)d * DS;
  #pragma unroll
  for (int n = 0; n < DS; ++n) acc += st[n] * w[n];
  out[(size_t)m * DM + d] = acc;
}

// ---------------------------------------------------------------- launch
extern "C" void kernel_launch(void* const* d_in, const int* in_sizes, int n_in,
                              void* d_out, int out_size, void* d_ws, size_t ws_size,
                              hipStream_t stream) {
  (void)in_sizes; (void)n_in; (void)out_size; (void)ws_size;
  const float* x       = (const float*)d_in[0];
  const float* W_delta = (const float*)d_in[1];
  const float* b_delta = (const float*)d_in[2];
  const float* W_B     = (const float*)d_in[3];
  const float* b_B     = (const float*)d_in[4];
  const float* A_log   = (const float*)d_in[5];
  const float* D_param = (const float*)d_in[6];
  const float* W_out   = (const float*)d_in[7];
  const float* b_out   = (const float*)d_in[8];
  float* out = (float*)d_out;

  char* ws = (char*)d_ws;
  size_t off = 0;
  auto alloc = [&](size_t bytes) -> void* {
    void* p = ws + off;
    off += (bytes + 255) & ~(size_t)255;
    return p;
  };
  _Float16* Xh    = (_Float16*)alloc((size_t)M_TOT * DM * sizeof(_Float16));
  _Float16* Wdh   = (_Float16*)alloc((size_t)DM * DM * sizeof(_Float16));
  _Float16* Wbh   = (_Float16*)alloc((size_t)DS * DM * sizeof(_Float16));
  float* xmean    = (float*)alloc((size_t)M_TOT * sizeof(float));
  float* dacc     = (float*)alloc((size_t)M_TOT * sizeof(float));
  float* Bm       = (float*)alloc((size_t)M_TOT * DS * sizeof(float));
  float* A_bar    = (float*)alloc((size_t)M_TOT * DS * sizeof(float));
  float* Bu       = (float*)alloc((size_t)M_TOT * DS * sizeof(float));
  float* states   = (float*)alloc((size_t)M_TOT * DS * sizeof(float));
  float* chunkP   = (float*)alloc((size_t)NCHAIN * NCHUNK * sizeof(float));
  float* chunkQ   = (float*)alloc((size_t)NCHAIN * NCHUNK * sizeof(float));
  float* chunkI   = (float*)alloc((size_t)NCHAIN * NCHUNK * sizeof(float));

  (void)hipMemsetAsync(dacc, 0, (size_t)M_TOT * sizeof(float), stream);

  cvt_f32_f16_kernel<<<(DM * DM) / 256, 256, 0, stream>>>(W_delta, Wdh, DM * DM);
  cvt_f32_f16_kernel<<<(DS * DM) / 256, 256, 0, stream>>>(W_B, Wbh, DS * DM);
  cvt_x_kernel<<<M_TOT, 256, 0, stream>>>(x, Xh, xmean);

  delta_gemm_kernel<<<dim3(DM / 64, M_TOT / 128), 256, 0, stream>>>(Xh, Wdh, b_delta, dacc);
  bm_gemm_kernel<<<M_TOT / 128, 256, 0, stream>>>(Xh, Wbh, b_B, Bm);

  abar_bu_kernel<<<(M_TOT * DS) / 256, 256, 0, stream>>>(dacc, Bm, xmean, A_log, A_bar, Bu);

  scan_phase1_kernel<<<(NCHAIN * NCHUNK) / 256, 256, 0, stream>>>(A_bar, Bu, chunkP, chunkQ);
  scan_phase2_kernel<<<1, NCHAIN, 0, stream>>>(chunkP, chunkQ, chunkI);
  scan_phase3_kernel<<<(NCHAIN * NCHUNK) / 256, 256, 0, stream>>>(A_bar, Bu, chunkI, states);

  out_gemm_kernel<<<M_TOT * 4, 256, 0, stream>>>(states, W_out, b_out, D_param, x, out);
}
